// JordanLSTM_32847909879809
// MI455X (gfx1250) — matrix operand order repaired and checked by reading, not compiled, not run on hardware
//
#include <hip/hip_runtime.h>
#include <hip/hip_bf16.h>
#include <math.h>

// ---------------------------------------------------------------------------
// Jordan LSTM  (B=64, T=512, I=512, H=1024, O=512)
//   Phase A (parallel):   Gx[t,b,:] = x[b,t,:] @ Wxh + bh          (WMMA bf16)
//   Phase B (sequential): per step t:
//       gates = Gx[t] + y @ Wyh        -> cell update -> h         (WMMA bf16)
//       y     = h @ Why + by           -> out[:,t,:], feedback     (WMMA bf16)
// A-tiles are staged in LDS (shared by all 8 waves of a WG) via
// GLOBAL_LOAD_ASYNC_TO_LDS_B128 (ASYNCcnt) with a ds_store fallback.
// ---------------------------------------------------------------------------

#define BB   64
#define TT   512
#define II   512
#define HH   1024
#define OO   512
#define G4H  4096   // 4*H
#define PAD  8      // bf16 elements of row padding in LDS (16B)

typedef __attribute__((ext_vector_type(16))) __bf16 v16bf;
typedef __attribute__((ext_vector_type(8)))  float  v8f;

// b128 async payload type: must match builtin's parameter (vector_size int4)
typedef int v4i_b128 __attribute__((vector_size(16)));
typedef __attribute__((address_space(1))) v4i_b128* async_gptr;
typedef __attribute__((address_space(3))) v4i_b128* async_lptr;

#if defined(__has_builtin)
#  if __has_builtin(__builtin_amdgcn_global_load_async_to_lds_b128)
#    define HAVE_ASYNC_LDS 1
#  else
#    define HAVE_ASYNC_LDS 0
#  endif
#else
#  define HAVE_ASYNC_LDS 0
#endif

#if defined(__has_builtin)
#  if __has_builtin(__builtin_amdgcn_s_wait_asynccnt)
#    define WAIT_ASYNC() __builtin_amdgcn_s_wait_asynccnt(0)
#  else
#    define WAIT_ASYNC() asm volatile("s_wait_asynccnt 0x0" ::: "memory")
#  endif
#else
#  define WAIT_ASYNC() asm volatile("s_wait_asynccnt 0x0" ::: "memory")
#endif

union FragAB {
    v16bf v;
    uint4 q[2];
};

// A fragment: 16x32 bf16 from a row-major tile with leading dim `lda`.
// lanes 0-15: K = [0..7],[16..23]; lanes 16-31: K = [8..15],[24..31]
__device__ __forceinline__ FragAB load_a_frag(const __hip_bfloat16* base,
                                              int lda, int row, int k0,
                                              int laneHi) {
    FragAB f;
    const __hip_bfloat16* p = base + (size_t)row * lda + k0 + (laneHi << 3);
    f.q[0] = *(const uint4*)(p);
    f.q[1] = *(const uint4*)(p + 16);
    return f;
}

// B fragment: 32x16 bf16 from TRANSPOSED weights (BT[n][k], K contiguous).
// lanes 0-15: K = [0..7],[16..23]; lanes 16-31: K = [8..15],[24..31]; column N = n0 + laneLo.
__device__ __forceinline__ FragAB load_bt_frag(const __hip_bfloat16* colptr,
                                               int k0, int laneHi) {
    FragAB f;
    const __hip_bfloat16* p = colptr + k0 + (laneHi << 3);
    f.q[0] = *(const uint4*)(p);
    f.q[1] = *(const uint4*)(p + 16);
    return f;
}

__device__ __forceinline__ v8f wmma_bf16(const FragAB& a, const FragAB& b, v8f c) {
    return __builtin_amdgcn_wmma_f32_16x16x32_bf16(
        false, a.v, false, b.v, (short)0, c, false, false);
}

__device__ __forceinline__ float sigmoid_f(float x) {
    return 1.0f / (1.0f + __expf(-x));
}

// Stage a 16-row x KDIM-col bf16 tile into LDS (row stride LDA, padded).
// 256 threads, 16-byte chunks; async-to-LDS when available.
template <int KDIM, int LDA, int CPT>
__device__ __forceinline__ void stage_tile(const __hip_bfloat16* __restrict__ g0,
                                           size_t grow_stride,
                                           __hip_bfloat16* sbase, int tid) {
    #pragma unroll
    for (int j = 0; j < CPT; ++j) {
        const int chunk = tid + 256 * j;                 // 16B chunk id
        const int row   = chunk / (KDIM / 8);
        const int col8  = (chunk % (KDIM / 8)) * 8;
        const __hip_bfloat16* src = g0 + (size_t)row * grow_stride + col8;
        __hip_bfloat16*       dst = sbase + row * LDA + col8;
#if HAVE_ASYNC_LDS
        __builtin_amdgcn_global_load_async_to_lds_b128(
            (async_gptr)(void*)src, (async_lptr)(void*)dst, 0, 0);
#else
        *(uint4*)dst = *(const uint4*)src;
#endif
    }
#if HAVE_ASYNC_LDS
    WAIT_ASYNC();
#endif
}

// Monotonic all-workgroup barrier (persistent kernel; counter only increases).
__device__ __forceinline__ void grid_barrier(unsigned* cnt, unsigned* gen) {
    __syncthreads();
    *gen += (unsigned)gridDim.x;
    if (threadIdx.x == 0) {
        __threadfence();
        atomicAdd(cnt, 1u);
        while (__hip_atomic_load(cnt, __ATOMIC_ACQUIRE,
                                 __HIP_MEMORY_SCOPE_AGENT) < *gen) {
            __builtin_amdgcn_s_sleep(8);
        }
    }
    __syncthreads();
}

// ---------------------------------------------------------------------------
// Setup kernels
// ---------------------------------------------------------------------------
__global__ void init_state_kernel(float* c_state, __hip_bfloat16* y_fb,
                                  unsigned* cnt) {
    unsigned i = blockIdx.x * blockDim.x + threadIdx.x;
    if (i == 0) *cnt = 0u;
    if (i < BB * HH) c_state[i] = 0.0f;
    if (i < BB * OO) y_fb[i] = __float2bfloat16(0.0f);
}

__global__ void cvt_f32_to_bf16_kernel(const float* __restrict__ in,
                                       __hip_bfloat16* __restrict__ out,
                                       unsigned long long n) {
    unsigned long long i =
        (unsigned long long)blockIdx.x * blockDim.x + threadIdx.x;
    if (i < n) out[i] = __float2bfloat16(in[i]);
}

// in: [rows][cols] fp32 -> out: [cols][rows] bf16
__global__ void transpose_to_bf16_kernel(const float* __restrict__ in,
                                         __hip_bfloat16* __restrict__ out,
                                         int rows, int cols) {
    unsigned long long i =
        (unsigned long long)blockIdx.x * blockDim.x + threadIdx.x;
    unsigned long long n = (unsigned long long)rows * cols;
    if (i < n) {
        int r = (int)(i / cols);
        int c = (int)(i % cols);
        out[(unsigned long long)c * rows + r] = __float2bfloat16(in[i]);
    }
}

// ---------------------------------------------------------------------------
// Phase A: Gx[t][b][n] = sum_k x[b][t][k] * Wxh[k][n] + bh[n]
// Block -> (mt, group of 8 nt); all 8 waves share one A-tile staged in LDS.
// M-space row R = t*64 + b  (so Gx is [T][B][4H]).
// ---------------------------------------------------------------------------
__global__ void __launch_bounds__(256)
gx_gemm_kernel(const __hip_bfloat16* __restrict__ xbf,   // [B][T][I]
               const __hip_bfloat16* __restrict__ WxhT,  // [4H][I]
               const float* __restrict__ bh,             // [4H]
               float* __restrict__ Gx) {                  // [T][B][4H]
    constexpr int LDA = II + PAD;
    __shared__ __hip_bfloat16 sX[16 * LDA];

    const int lane   = threadIdx.x & 31;
    const int wave   = threadIdx.x >> 5;
    const int laneLo = lane & 15;
    const int laneHi = lane >> 4;

    const int mt = blockIdx.x >> 5;                  // 0..2047 (shared by WG)
    const int nt = (blockIdx.x & 31) * 8 + wave;     // 0..255
    const int m0 = mt * 16;
    const int n0 = nt * 16;
    const int b0 = m0 & 63;                          // t constant in tile
    const int t0 = m0 >> 6;

    // cooperative A-tile stage: rows r -> x row (b0+r, t0)
    stage_tile<II, LDA, 4>(xbf + ((size_t)b0 * TT + t0) * II,
                           (size_t)TT * II, sX, threadIdx.x);
    __syncthreads();

    const int ncol = n0 + laneLo;
    const __hip_bfloat16* bcol = WxhT + (size_t)ncol * II;

    v8f acc = {};
    #pragma unroll 4
    for (int k0 = 0; k0 < II; k0 += 32) {
        FragAB a  = load_a_frag(sX, LDA, laneLo, k0, laneHi);
        FragAB bb = load_bt_frag(bcol, k0, laneHi);
        acc = wmma_bf16(a, bb, acc);
    }

    const float bias  = bh[ncol];
    const int   mbase = m0 + laneHi * 8;
    #pragma unroll
    for (int r = 0; r < 8; ++r) {
        const int RR = mbase + r;
        const int br = RR & 63;
        const int tr = RR >> 6;
        Gx[((size_t)tr * BB + br) * G4H + ncol] = acc[r] + bias;
    }
}

// ---------------------------------------------------------------------------
// Phase B: persistent sequential kernel. 32 WGs x 8 waves = 256 waves.
// Phase 1: wave u (mb=u>>6, hb=u&63) computes gate tiles {i,f,g,o} at
//          rows [mb*16,+16), cols hb*16 + g*1024 -> cell update in registers.
//          mb is uniform per WG -> A-tile (y_fb) staged once in LDS.
// Phase 2: wave u<128 (mb2=u>>5, nb2=u&31): y tile; mb2 uniform per WG ->
//          A-tile (h_buf) staged once in LDS.
// ---------------------------------------------------------------------------
__global__ void __launch_bounds__(256, 1)
lstm_seq_kernel(const float* __restrict__ Gx,             // [T][B][4H]
                const __hip_bfloat16* __restrict__ WyhT,  // [4H][O]
                const __hip_bfloat16* __restrict__ WhyT,  // [O][H]
                const float* __restrict__ by,             // [O]
                float* __restrict__ c_state,              // [B][H]
                __hip_bfloat16* __restrict__ y_fb,        // [B][O]
                __hip_bfloat16* __restrict__ h_buf,       // [B][H]
                unsigned* __restrict__ bar,
                float* __restrict__ out) {                 // [B][T][O]
    constexpr int LDA1 = OO + PAD;   // phase-1 tile leading dim
    constexpr int LDA2 = HH + PAD;   // phase-2 tile leading dim
    __shared__ __hip_bfloat16 sA[16 * LDA2];   // 33 KB, reused per phase

    const int lane   = threadIdx.x & 31;
    const int wave   = threadIdx.x >> 5;
    const int gwave  = blockIdx.x * 8 + wave;   // 0..255
    const int laneLo = lane & 15;
    const int laneHi = lane >> 4;

    // phase-1 unit (mb uniform within a WG)
    const int mb    = gwave >> 6;               // 0..3 (b tile)
    const int hb    = gwave & 63;               // 0..63 (h column block)
    const int m0    = mb * 16;
    const int mbase = m0 + laneHi * 8;
    const int hcol  = hb * 16 + laneLo;         // 0..1023

    // phase-2 unit (valid for gwave < 128; mb2 uniform within a WG)
    const int mb2    = gwave >> 5;
    const int nb2    = gwave & 31;
    const int m0_2   = mb2 * 16;
    const int mbase2 = m0_2 + laneHi * 8;
    const int ncol2  = nb2 * 16 + laneLo;       // 0..511

    unsigned gen = 0;

    for (int t = 0; t < TT; ++t) {
        // ---------- Phase 1: gates = Gx[t] + y_fb @ Wyh ----------
        stage_tile<OO, LDA1, 4>(y_fb + (size_t)m0 * OO, (size_t)OO, sA,
                                threadIdx.x);
        __syncthreads();

        const float* gxt = Gx + (size_t)t * (BB * G4H);
        v8f acc[4];
        #pragma unroll
        for (int g = 0; g < 4; ++g) {
            #pragma unroll
            for (int r = 0; r < 8; ++r)
                acc[g][r] = gxt[(size_t)(mbase + r) * G4H + g * HH + hcol];
        }

        #pragma unroll 1
        for (int k0 = 0; k0 < OO; k0 += 32) {
            FragAB a = load_a_frag(sA, LDA1, laneLo, k0, laneHi);
            #pragma unroll
            for (int g = 0; g < 4; ++g) {
                const __hip_bfloat16* bcol =
                    WyhT + (size_t)(g * HH + hcol) * OO;
                FragAB b = load_bt_frag(bcol, k0, laneHi);
                acc[g] = wmma_bf16(a, b, acc[g]);
            }
        }

        // cell update fully in registers: this lane owns (b=mbase+r, h=hcol)
        #pragma unroll
        for (int r = 0; r < 8; ++r) {
            const int    brow = mbase + r;
            const size_t ch   = (size_t)brow * HH + hcol;
            const float  cprev = c_state[ch];
            const float  ig = sigmoid_f(acc[0][r]);
            const float  fg = sigmoid_f(acc[1][r]);
            const float  gg = tanhf(acc[2][r]);
            const float  og = sigmoid_f(acc[3][r]);
            const float  cn = fg * cprev + ig * gg;
            c_state[ch] = cn;
            h_buf[ch]   = __float2bfloat16(og * tanhf(cn));
        }

        grid_barrier(bar, &gen);   // h complete everywhere

        // ---------- Phase 2: y = h @ Why + by ----------
        if (blockIdx.x < 16) {     // uniform per block; gwave<128 here
            stage_tile<HH, LDA2, 8>(h_buf + (size_t)m0_2 * HH, (size_t)HH, sA,
                                    threadIdx.x);
            __syncthreads();

            v8f acc2;
            const float bias = by[ncol2];
            #pragma unroll
            for (int r = 0; r < 8; ++r) acc2[r] = bias;

            const __hip_bfloat16* bcol2 = WhyT + (size_t)ncol2 * HH;
            #pragma unroll 2
            for (int k0 = 0; k0 < HH; k0 += 32) {
                FragAB a = load_a_frag(sA, LDA2, laneLo, k0, laneHi);
                FragAB b = load_bt_frag(bcol2, k0, laneHi);
                acc2 = wmma_bf16(a, b, acc2);
            }
            #pragma unroll
            for (int r = 0; r < 8; ++r) {
                const int   brow = mbase2 + r;
                const float v    = acc2[r];
                out [((size_t)brow * TT + t) * OO + ncol2] = v;
                y_fb[ (size_t)brow * OO + ncol2]           = __float2bfloat16(v);
            }
        } else if ((t + 1) < TT && lane == 0) {
            // idle waves prefetch next step's Gx block (global_prefetch_b8)
            const float* nxt = Gx + (size_t)(t + 1) * (BB * G4H)
                                  + (size_t)(gwave - 128) * (BB * G4H / 128);
            __builtin_prefetch((const void*)nxt, 0, 1);
        }

        grid_barrier(bar, &gen);   // y feedback complete everywhere
    }
}

// ---------------------------------------------------------------------------
// Host launcher
// ---------------------------------------------------------------------------
extern "C" void kernel_launch(void* const* d_in, const int* in_sizes, int n_in,
                              void* d_out, int out_size, void* d_ws,
                              size_t ws_size, hipStream_t stream) {
    (void)in_sizes; (void)n_in; (void)out_size; (void)ws_size;

    const float* x   = (const float*)d_in[0];   // [B][T][I]
    const float* Wxh = (const float*)d_in[1];   // [I][4H]
    const float* Wyh = (const float*)d_in[2];   // [O][4H]
    const float* bh  = (const float*)d_in[3];   // [4H]
    const float* by  = (const float*)d_in[4];   // [O]
    const float* Why = (const float*)d_in[5];   // [H][O]
    float* out = (float*)d_out;                 // [B][T][O]

    // workspace carve (256B aligned)
    char*  ws  = (char*)d_ws;
    size_t off = 0;
    auto carve = [&](size_t bytes) -> void* {
        off = (off + 255) & ~(size_t)255;
        void* p = ws + off;
        off += bytes;
        return p;
    };
    unsigned*       bar     = (unsigned*)      carve(256);
    float*          c_state = (float*)         carve((size_t)BB * HH * 4);
    __hip_bfloat16* y_fb    = (__hip_bfloat16*)carve((size_t)BB * OO * 2);
    __hip_bfloat16* h_buf   = (__hip_bfloat16*)carve((size_t)BB * HH * 2);
    __hip_bfloat16* WyhT    = (__hip_bfloat16*)carve((size_t)G4H * OO * 2);
    __hip_bfloat16* WhyT    = (__hip_bfloat16*)carve((size_t)OO * HH * 2);
    __hip_bfloat16* WxhT    = (__hip_bfloat16*)carve((size_t)G4H * II * 2);
    __hip_bfloat16* xbf     = (__hip_bfloat16*)carve((size_t)BB * TT * II * 2);
    float*          Gx      = (float*)         carve((size_t)TT * BB * G4H * 4);

    // 1) init persistent state
    init_state_kernel<<<(BB * HH + 255) / 256, 256, 0, stream>>>(c_state, y_fb, bar);

    // 2) precision conversions / weight transposes
    {
        unsigned long long n = (unsigned long long)BB * TT * II;
        cvt_f32_to_bf16_kernel<<<(unsigned)((n + 255) / 256), 256, 0, stream>>>(x, xbf, n);
    }
    transpose_to_bf16_kernel<<<(II * G4H + 255) / 256, 256, 0, stream>>>(Wxh, WxhT, II, G4H);
    transpose_to_bf16_kernel<<<(OO * G4H + 255) / 256, 256, 0, stream>>>(Wyh, WyhT, OO, G4H);
    transpose_to_bf16_kernel<<<(HH * OO  + 255) / 256, 256, 0, stream>>>(Why, WhyT, HH, OO);

    // 3) Phase A: big parallel WMMA GEMM  Gx = x@Wxh + bh
    //    blocks: 2048 mt * 32 nt-groups; 8 waves/WG share one LDS A-tile
    gx_gemm_kernel<<<2048 * 32, 256, 0, stream>>>(xbf, WxhT, bh, Gx);

    // 4) Phase B: persistent sequential recurrence (32 WGs, grid barrier)
    lstm_seq_kernel<<<32, 256, 0, stream>>>(Gx, WyhT, WhyT, by, c_state, y_fb,
                                            h_buf, bar, out);
}
